// Transformer_DE_6493990552306
// MI455X (gfx1250) — matrix operand order repaired and checked by reading, not compiled, not run on hardware
//
#include <hip/hip_runtime.h>
#include <hip/hip_bf16.h>

typedef __attribute__((ext_vector_type(16))) __bf16 v16bf;
typedef __attribute__((ext_vector_type(8)))  float  v8f;
typedef __attribute__((ext_vector_type(4)))  unsigned int tdm_v4u;
typedef __attribute__((ext_vector_type(8)))  int          tdm_v8i;
typedef __attribute__((ext_vector_type(4)))  int          tdm_v4i;

#define Bn 4
#define Ln 640
#define Dm 480
#define Hh 2
#define DKh 960
#define DKH 1920
#define BL 2560
#define KNUM 10

// ---------------- device helpers ----------------
__device__ __forceinline__ unsigned short f2bf(float f) {
  unsigned u = __float_as_uint(f);
  unsigned r = (u + 0x7FFFu + ((u >> 16) & 1u)) >> 16;
  return (unsigned short)r;
}
__device__ __forceinline__ unsigned pack2(float lo, float hi) {
  return (unsigned)f2bf(lo) | ((unsigned)f2bf(hi) << 16);
}
__device__ __forceinline__ float gelu_f(float x) {
  return 0.5f * x * (1.f + erff(x * 0.70710678118654752f));
}
__device__ __forceinline__ float selu_f(float x) {
  const float a = 1.6732632423543772f, s = 1.0507009873554805f;
  return x > 0.f ? s * x : s * a * (__expf(x) - 1.f);
}
// Gaussian-RBF expand (K=10) -> 10x10 MLP (gelu) -> 10x1
__device__ float phi_dev(float x, const float* mu, const float* sg, const float* bb,
                         const float* w1, const float* b1, const float* w2, float b2out) {
  float psi[KNUM];
#pragma unroll
  for (int k = 0; k < KNUM; ++k) {
    float s = sg[k];
    float t = (x + bb[k] - mu[k]) / s;
    psi[k] = __expf(-0.5f * t * t) / (2.5066282746310002f * s);
  }
  float out = b2out;
#pragma unroll
  for (int j = 0; j < KNUM; ++j) {
    float hj = b1[j];
#pragma unroll
    for (int k = 0; k < KNUM; ++k) hj += psi[k] * w1[k * KNUM + j];
    out += gelu_f(hj) * w2[j];
  }
  return out;
}

// ---- Tensor Data Mover: 1D row copy (global f32 -> LDS), D# per CDNA5 ISA §8 ----
__device__ __forceinline__ void tdm_copy_row(const float* src, unsigned lds_byte_off, int nelem) {
  unsigned long long ga = (unsigned long long)(const void*)src;
  tdm_v4u g0;
  g0.x = 1u;                                   // count=1 (valid user descriptor)
  g0.y = lds_byte_off;                         // lds_addr
  g0.z = (unsigned)(ga & 0xFFFFFFFFu);         // global_addr[31:0]
  g0.w = (unsigned)((ga >> 32) & 0x1FFFFFFu) | (2u << 30);  // global_addr[56:32] | type=2
  tdm_v8i g1;
  unsigned ne = (unsigned)nelem;
  g1[0] = (int)(2u << 16);                     // data_size = 4 bytes
  g1[1] = (int)(ne << 16);                     // tensor_dim0[15:0] in bits[31:16]
  g1[2] = (int)((ne >> 16) | (1u << 16));      // tensor_dim0[31:16]; tensor_dim1 = 1
  g1[3] = (int)(ne << 16);                     // tensor_dim1 hi = 0; tile_dim0 = nelem
  g1[4] = 1;                                   // tile_dim1 = 1, tile_dim2 = 0
  g1[5] = (int)ne;                             // tensor_dim0_stride[31:0]
  g1[6] = 0;
  g1[7] = 0;
  tdm_v4i z4 = {0, 0, 0, 0};
  tdm_v8i z8 = {0, 0, 0, 0, 0, 0, 0, 0};
  __builtin_amdgcn_tensor_load_to_lds(g0, g1, z4, z4, z8, 0);
}

// ---------------- batched WMMA GEMM (bf16 in, f32 acc) ----------------
// C[z] = epi( (A[z] @ B[z](^T if BT)) * scale + bias ) + resid
// offsets: off = (z/div)*s1 + (z%div)*s2
#define KPAD 36
template <bool BT, int EPI>
__global__ __launch_bounds__(128) void gemm_wmma(
    const float* __restrict__ A, const float* __restrict__ Bm,
    const float* __restrict__ bias, const float* __restrict__ resid,
    float* __restrict__ C, int M, int N, int K, int lda, int ldb, int ldc,
    float scale, int div, long sA1, long sA2, long sB1, long sB2, long sC1, long sC2) {
  __shared__ __align__(16) unsigned short As[64 * KPAD];
  __shared__ __align__(16) unsigned short Bs[64 * KPAD];

  const int tid = threadIdx.x;
  const int lane = tid & 31;
  const int wave = tid >> 5;
  const int wm = (wave >> 1) * 32, wn = (wave & 1) * 32;
  const int l16 = lane & 15, hi = lane >> 4;
  const int z = blockIdx.z;
  const long offA = (long)(z / div) * sA1 + (long)(z % div) * sA2;
  const long offB = (long)(z / div) * sB1 + (long)(z % div) * sB2;
  const long offC = (long)(z / div) * sC1 + (long)(z % div) * sC2;
  const int m0 = blockIdx.y * 64, n0 = blockIdx.x * 64;
  const bool fullM = (m0 + 64 <= M);
  const bool fullN = (n0 + 64 <= N);

  v8f acc[2][2];
#pragma unroll
  for (int i = 0; i < 2; ++i)
#pragma unroll
    for (int j = 0; j < 2; ++j)
#pragma unroll
      for (int r = 0; r < 8; ++r) acc[i][j][r] = 0.f;

  for (int kb = 0; kb < K; kb += 32) {
    const bool fullK = (kb + 32 <= K);
    // ---- stage A tile (64 x 32) as bf16, K-contiguous ----
    if (fullM && fullK) {
#pragma unroll
      for (int i = tid; i < 512; i += 128) {  // 512 float4 chunks
        int r = i >> 3, c4 = (i & 7) * 4;
        const float4 f = *(const float4*)&A[offA + (long)(m0 + r) * lda + kb + c4];
        *(uint2*)&As[r * KPAD + c4] = make_uint2(pack2(f.x, f.y), pack2(f.z, f.w));
      }
    } else {
      for (int i = tid; i < 64 * 32; i += 128) {
        int r = i >> 5, c = i & 31;
        int gm = m0 + r, gk = kb + c;
        float v = 0.f;
        if (gm < M && gk < K) v = A[offA + (long)gm * lda + gk];
        As[r * KPAD + c] = f2bf(v);
      }
    }
    // ---- stage B tile as Bt[n][k] (K-contiguous in LDS) ----
    if (BT) {
      if (fullN && fullK) {
#pragma unroll
        for (int i = tid; i < 512; i += 128) {
          int r = i >> 3, c4 = (i & 7) * 4;
          const float4 f = *(const float4*)&Bm[offB + (long)(n0 + r) * ldb + kb + c4];
          *(uint2*)&Bs[r * KPAD + c4] = make_uint2(pack2(f.x, f.y), pack2(f.z, f.w));
        }
      } else {
        for (int i = tid; i < 64 * 32; i += 128) {
          int r = i >> 5, c = i & 31;  // r=n, c=k
          int gn = n0 + r, gk = kb + c;
          float v = 0.f;
          if (gn < N && gk < K) v = Bm[offB + (long)gn * ldb + gk];
          Bs[r * KPAD + c] = f2bf(v);
        }
      }
    } else {
      if (fullN && fullK) {
#pragma unroll
        for (int i = tid; i < 256; i += 128) {  // (32/2 k-pairs) x (64/4 n-quads)
          int kp = i >> 4, n4 = (i & 15) * 4;
          int k = kp * 2;
          const float4 r0 = *(const float4*)&Bm[offB + (long)(kb + k) * ldb + n0 + n4];
          const float4 r1 = *(const float4*)&Bm[offB + (long)(kb + k + 1) * ldb + n0 + n4];
          *(unsigned*)&Bs[(n4 + 0) * KPAD + k] = pack2(r0.x, r1.x);
          *(unsigned*)&Bs[(n4 + 1) * KPAD + k] = pack2(r0.y, r1.y);
          *(unsigned*)&Bs[(n4 + 2) * KPAD + k] = pack2(r0.z, r1.z);
          *(unsigned*)&Bs[(n4 + 3) * KPAD + k] = pack2(r0.w, r1.w);
        }
      } else {
        for (int i = tid; i < 32 * 64; i += 128) {
          int r = i >> 6, c = i & 63;  // r=k, c=n (coalesced)
          int gk = kb + r, gn = n0 + c;
          float v = 0.f;
          if (gk < K && gn < N) v = Bm[offB + (long)gk * ldb + gn];
          Bs[c * KPAD + r] = f2bf(v);
        }
      }
    }
    __syncthreads();

    union Frag { v16bf v; unsigned u[8]; };
    Frag af[2], bf[2];
#pragma unroll
    for (int t = 0; t < 2; ++t) {
      const int mrow = wm + t * 16 + l16;
#pragma unroll
      for (int j = 0; j < 8; ++j) {
        int kk = ((j < 4) ? (2 * j) : (16 + 2 * (j - 4))) + hi * 8;
        af[t].u[j] = *(const unsigned*)&As[mrow * KPAD + kk];
      }
      const int nrow = wn + t * 16 + l16;
#pragma unroll
      for (int j = 0; j < 8; ++j) {
        int kk = ((j < 4) ? (2 * j) : (16 + 2 * (j - 4))) + hi * 8;
        bf[t].u[j] = *(const unsigned*)&Bs[nrow * KPAD + kk];
      }
    }
#pragma unroll
    for (int i = 0; i < 2; ++i)
#pragma unroll
      for (int j = 0; j < 2; ++j)
        acc[i][j] = __builtin_amdgcn_wmma_f32_16x16x32_bf16(
            false, af[i].v, false, bf[j].v, (short)0, acc[i][j], false, false);
    __syncthreads();
  }

#pragma unroll
  for (int i = 0; i < 2; ++i)
#pragma unroll
    for (int j = 0; j < 2; ++j)
#pragma unroll
      for (int r = 0; r < 8; ++r) {
        int row = m0 + wm + i * 16 + (hi ? 8 : 0) + r;
        int col = n0 + wn + j * 16 + l16;
        if (row < M && col < N) {
          float v = acc[i][j][r] * scale;
          if (bias) v += bias[col];
          if (EPI == 1) v = gelu_f(v);
          if (EPI == 2) v = selu_f(v);
          if (resid) v += resid[offC + (long)row * ldc + col];
          C[offC + (long)row * ldc + col] = v;
        }
      }
}

// ---------------- fused RBF-bias + mask + softmax over attention rows ----------------
template <int NPHI>
__global__ __launch_bounds__(256) void attn_softmax(
    float* __restrict__ scores, const float* __restrict__ dist,
    const float* __restrict__ energy, const int* __restrict__ mask,
    const float* muD, const float* sgD, const float* bbD,
    const float* muE, const float* sgE, const float* bbE,
    const float* W1w, const float* W1b, const float* W2w, const float* W2b) {
  __shared__ float rowbuf[Ln];
  __shared__ float red[256];
  __shared__ float p[181];
  const int tid = threadIdx.x;
  if (tid < KNUM) {
    p[tid] = muD[tid]; p[10 + tid] = sgD[tid]; p[20 + tid] = bbD[tid];
    p[160 + tid] = W1b[tid]; p[170 + tid] = W2w[tid];
    if (NPHI == 2) { p[30 + tid] = muE[tid]; p[40 + tid] = sgE[tid]; p[50 + tid] = bbE[tid]; }
  }
  if (tid < 100) p[60 + tid] = W1w[tid];
  if (tid == 0) p[180] = W2b[0];
  __syncthreads();

  const int qrow = blockIdx.x, z = blockIdx.y, b = z >> 1;
  const long sbase = ((long)z * Ln + qrow) * Ln;
  const long dbase = ((long)b * Ln + qrow) * Ln;
  float lmax = -3.4e38f;
  for (int c = tid; c < Ln; c += 256) {
    float s = scores[sbase + c];
    s += phi_dev(dist[dbase + c], p, p + 10, p + 20, p + 60, p + 160, p + 170, p[180]);
    if (NPHI == 2)
      s += phi_dev(energy[dbase + c], p + 30, p + 40, p + 50, p + 60, p + 160, p + 170, p[180]);
    if (mask[dbase + c] == 0) s = -1e9f;
    rowbuf[c] = s;
    lmax = fmaxf(lmax, s);
  }
  red[tid] = lmax; __syncthreads();
  for (int st = 128; st > 0; st >>= 1) { if (tid < st) red[tid] = fmaxf(red[tid], red[tid + st]); __syncthreads(); }
  float mx = red[0]; __syncthreads();
  float lsum = 0.f;
  for (int c = tid; c < Ln; c += 256) { float e = __expf(rowbuf[c] - mx); rowbuf[c] = e; lsum += e; }
  red[tid] = lsum; __syncthreads();
  for (int st = 128; st > 0; st >>= 1) { if (tid < st) red[tid] += red[tid + st]; __syncthreads(); }
  float inv = 1.f / red[0];
  for (int c = tid; c < Ln; c += 256) scores[sbase + c] = rowbuf[c] * inv;
}

// ---------------- layernorm ----------------
__global__ __launch_bounds__(256) void layernorm_k(const float* __restrict__ x,
                                                   const float* __restrict__ g,
                                                   const float* __restrict__ bb,
                                                   float* __restrict__ y, int D) {
  __shared__ float r1[256], r2[256];
  const int row = blockIdx.x, tid = threadIdx.x;
  const float* xr = x + (long)row * D;
  float s = 0.f, ss = 0.f;
  for (int i = tid; i < D; i += 256) { float v = xr[i]; s += v; ss += v * v; }
  r1[tid] = s; r2[tid] = ss; __syncthreads();
  for (int st = 128; st > 0; st >>= 1) {
    if (tid < st) { r1[tid] += r1[tid + st]; r2[tid] += r2[tid + st]; }
    __syncthreads();
  }
  float mean = r1[0] / D;
  float var = r2[0] / D - mean * mean;
  float inv = rsqrtf(var + 1e-5f);
  for (int i = tid; i < D; i += 256) y[(long)row * D + i] = (xr[i] - mean) * inv * g[i] + bb[i];
}

// ---------------- small row-dot (K->1), optional sigmoid ----------------
template <int ACT>
__global__ void rowdot_k(const float* __restrict__ h, const float* __restrict__ w,
                         const float* __restrict__ bias, float* __restrict__ out, int n, int K) {
  int r = blockIdx.x * blockDim.x + threadIdx.x;
  if (r >= n) return;
  float acc = bias[0];
  for (int k = 0; k < K; ++k) acc += h[(long)r * K + k] * w[k];
  if (ACT == 3) acc = 1.f / (1.f + __expf(-acc));
  out[r] = acc;
}

// ---------------- per-row activation RBF-MLP (asf -> psi) ----------------
__global__ void act_psi_k(const float* __restrict__ asf, const float* mu, const float* sg,
                          const float* bb, const float* w1, const float* b1, const float* w2,
                          const float* b2, float* __restrict__ out, int n) {
  int r = blockIdx.x * blockDim.x + threadIdx.x;
  if (r >= n) return;
  out[r] = phi_dev(asf[r], mu, sg, bb, w1, b1, w2, b2[0]);
}

// ---------------- pooling softmax weights per batch ----------------
__global__ __launch_bounds__(256) void pool_w_k(const float* __restrict__ psi,
                                                const int* __restrict__ mask,
                                                float* __restrict__ w) {
  __shared__ float buf[Ln];
  __shared__ float red[256];
  const int b = blockIdx.x, tid = threadIdx.x;
  float lmax = -3.4e38f;
  for (int l = tid; l < Ln; l += 256) {
    float m = (float)mask[(long)b * Ln * Ln + (long)l * Ln];
    float v = psi[b * Ln + l] * m;
    if (v == 0.f) v = -1e9f;
    buf[l] = v; lmax = fmaxf(lmax, v);
  }
  red[tid] = lmax; __syncthreads();
  for (int st = 128; st > 0; st >>= 1) { if (tid < st) red[tid] = fmaxf(red[tid], red[tid + st]); __syncthreads(); }
  float mx = red[0]; __syncthreads();
  float lsum = 0.f;
  for (int l = tid; l < Ln; l += 256) { float e = __expf(buf[l] - mx); buf[l] = e; lsum += e; }
  red[tid] = lsum; __syncthreads();
  for (int st = 128; st > 0; st >>= 1) { if (tid < st) red[tid] += red[tid + st]; __syncthreads(); }
  float inv = 1.f / red[0];
  for (int l = tid; l < Ln; l += 256) w[b * Ln + l] = buf[l] * inv;
}

__global__ void pool_e_k(const float* __restrict__ w, const float* __restrict__ e,
                         float* __restrict__ pooled, int n) {
  int t = blockIdx.x * blockDim.x + threadIdx.x;
  if (t >= n) return;
  int b = t / Dm, dcol = t % Dm;
  float acc = 0.f;
  for (int l = 0; l < Ln; ++l) acc += w[b * Ln + l] * e[((long)b * Ln + l) * Dm + dcol];
  pooled[t] = acc;
}

// ---------------- conv1d (K=31, pad 15) with TDM row staging ----------------
__global__ void conv_wt_k(const float* __restrict__ w, float* __restrict__ wt, int n) {
  int i = blockIdx.x * blockDim.x + threadIdx.x;
  if (i >= n) return;
  int oc = i / (DKh * 31);
  int rem = i % (DKh * 31);
  int ic = rem / 31, k = rem % 31;
  wt[((long)k * DKh + ic) * 128 + oc] = w[i];
}

__global__ __launch_bounds__(128) void conv1d_k(const float* __restrict__ s,
                                                const float* __restrict__ pooled,
                                                const float* __restrict__ wt,
                                                const float* __restrict__ cb,
                                                float* __restrict__ out) {
  __shared__ __align__(16) float xrow[DKh];
  const int t = blockIdx.x, b = blockIdx.y, oc = threadIdx.x;
  const unsigned lds0 = (unsigned)(unsigned long long)(void*)&xrow[0];
  // pooled half of the channel vector: constant across taps, copy once via TDM
  if (oc < 32) tdm_copy_row(pooled + b * Dm, lds0 + Dm * 4u, Dm);
  float acc = cb[oc];
  for (int k = 0; k < 31; ++k) {
    int tin = t + k - 15;
    bool ok = (tin >= 0) && (tin < Ln);
    __syncthreads();
    if (ok && oc < 32) tdm_copy_row(s + ((long)b * Ln + tin) * Dm, lds0, Dm);
    if (oc < 32) __builtin_amdgcn_s_wait_tensorcnt(0);
    __syncthreads();
    if (ok) {
      const float* wp = wt + (long)k * DKh * 128 + oc;
      for (int ic = 0; ic < DKh; ++ic) acc += xrow[ic] * wp[ic * 128];
    }
  }
  out[((long)b * Ln + t) * 128 + oc] = selu_f(acc);
}

// ---------------- host side ----------------
static inline dim3 ggrid(int M, int N, int batch) {
  return dim3((unsigned)((N + 63) / 64), (unsigned)((M + 63) / 64), (unsigned)batch);
}

struct EncBufs {
  float *q, *k, *v, *scores, *ctx, *ypre, *y, *hffn;
};

static void run_encoder(hipStream_t st, const float* x, const float* dist, const float* energy,
                        const int* mask, const float* Wqw, const float* Wqb, const float* Wkw,
                        const float* Wkb, const float* Wvw, const float* Wvb, const float* Wow,
                        const float* Wob, const float* f1w, const float* f1b, const float* f2w,
                        const float* f2b, const float* lng, const float* lnb, const float* muD,
                        const float* sgD, const float* bbD, const float* muE, const float* sgE,
                        const float* bbE, const float* WD1w, const float* WD1b, const float* WD2w,
                        const float* WD2b, EncBufs bufs, float* out) {
  const float iscale = 0.03227486121839514f;  // 1/sqrt(960)
  gemm_wmma<false, 0><<<ggrid(BL, DKH, 1), 128, 0, st>>>(x, Wqw, Wqb, nullptr, bufs.q, BL, DKH, Dm,
                                                         Dm, DKH, DKH, 1.f, 1, 0, 0, 0, 0, 0, 0);
  gemm_wmma<false, 0><<<ggrid(BL, DKH, 1), 128, 0, st>>>(x, Wkw, Wkb, nullptr, bufs.k, BL, DKH, Dm,
                                                         Dm, DKH, DKH, 1.f, 1, 0, 0, 0, 0, 0, 0);
  gemm_wmma<false, 0><<<ggrid(BL, DKH, 1), 128, 0, st>>>(x, Wvw, Wvb, nullptr, bufs.v, BL, DKH, Dm,
                                                         Dm, DKH, DKH, 1.f, 1, 0, 0, 0, 0, 0, 0);
  gemm_wmma<true, 0><<<ggrid(Ln, Ln, Bn * Hh), 128, 0, st>>>(
      bufs.q, bufs.k, nullptr, nullptr, bufs.scores, Ln, Ln, DKh, DKH, DKH, Ln, iscale, Hh,
      (long)Ln * DKH, DKh, (long)Ln * DKH, DKh, (long)Hh * Ln * Ln, (long)Ln * Ln);
  if (energy)
    attn_softmax<2><<<dim3(Ln, Bn * Hh), 256, 0, st>>>(bufs.scores, dist, energy, mask, muD, sgD,
                                                       bbD, muE, sgE, bbE, WD1w, WD1b, WD2w, WD2b);
  else
    attn_softmax<1><<<dim3(Ln, Bn * Hh), 256, 0, st>>>(bufs.scores, dist, nullptr, mask, muD, sgD,
                                                       bbD, nullptr, nullptr, nullptr, WD1w, WD1b,
                                                       WD2w, WD2b);
  gemm_wmma<false, 0><<<ggrid(Ln, DKh, Bn * Hh), 128, 0, st>>>(
      bufs.scores, bufs.v, nullptr, nullptr, bufs.ctx, Ln, DKh, Ln, Ln, DKH, DKH, 1.f, Hh,
      (long)Hh * Ln * Ln, (long)Ln * Ln, (long)Ln * DKH, DKh, (long)Ln * DKH, DKh);
  gemm_wmma<false, 0><<<ggrid(BL, Dm, 1), 128, 0, st>>>(bufs.ctx, Wow, Wob, x, bufs.ypre, BL, Dm,
                                                        DKH, DKH, Dm, Dm, 1.f, 1, 0, 0, 0, 0, 0, 0);
  layernorm_k<<<BL, 256, 0, st>>>(bufs.ypre, lng, lnb, bufs.y, Dm);
  gemm_wmma<false, 1><<<ggrid(BL, 4 * Dm, 1), 128, 0, st>>>(bufs.y, f1w, f1b, nullptr, bufs.hffn,
                                                            BL, 4 * Dm, Dm, Dm, 4 * Dm, 4 * Dm, 1.f,
                                                            1, 0, 0, 0, 0, 0, 0);
  gemm_wmma<false, 0><<<ggrid(BL, Dm, 1), 128, 0, st>>>(bufs.hffn, f2w, f2b, nullptr, out, BL, Dm,
                                                        4 * Dm, 4 * Dm, Dm, Dm, 1.f, 1, 0, 0, 0, 0,
                                                        0, 0);
}

extern "C" void kernel_launch(void* const* d_in, const int* in_sizes, int n_in, void* d_out,
                              int out_size, void* d_ws, size_t ws_size, hipStream_t stream) {
  (void)in_sizes; (void)n_in; (void)out_size; (void)ws_size;
  #define FIN(i) ((const float*)d_in[(i)])
  const float* subF = FIN(0);
  const float* subD = FIN(1);
  const int*   subM = (const int*)d_in[2];
  const float* enzF = FIN(3);
  const float* enzD = FIN(4);
  const float* enzE = FIN(5);
  const int*   enzM = (const int*)d_in[6];
  // params leaves (jax pytree: dict keys sorted, lin = {b,w})
  const float *cls1b = FIN(7),  *cls1w = FIN(8);
  const float *cls2b = FIN(9),  *cls2w = FIN(10);
  const float *cls3b = FIN(11), *cls3w = FIN(12);
  const float *convb = FIN(13), *convw = FIN(14);
  const float *eWkb = FIN(15), *eWkw = FIN(16), *eWob = FIN(17), *eWow = FIN(18);
  const float *eWqb = FIN(19), *eWqw = FIN(20), *eWvb = FIN(21), *eWvw = FIN(22);
  const float *ef1b = FIN(23), *ef1w = FIN(24), *ef2b = FIN(25), *ef2w = FIN(26);
  const float *elnb = FIN(27), *elng = FIN(28);
  const float *fc1b = FIN(29), *fc1w = FIN(30), *fc2b = FIN(31), *fc2w = FIN(32);
  const float *WD1b = FIN(33), *WD1w = FIN(34), *WD2b = FIN(35), *WD2w = FIN(36);
  const float *Wa1b = FIN(37), *Wa1w = FIN(38), *Wa2b = FIN(39), *Wa2w = FIN(40);
  const float *bDE = FIN(41), *bDS = FIN(42), *bE = FIN(43), *bACT = FIN(44);
  const float *muDE = FIN(45), *muDS = FIN(46), *muE = FIN(47), *muACT = FIN(48);
  const float *sgDE = FIN(49), *sgDS = FIN(50), *sgE = FIN(51), *sgACT = FIN(52);
  const float *sWkb = FIN(53), *sWkw = FIN(54), *sWob = FIN(55), *sWow = FIN(56);
  const float *sWqb = FIN(57), *sWqw = FIN(58), *sWvb = FIN(59), *sWvw = FIN(60);
  const float *sf1b = FIN(61), *sf1w = FIN(62), *sf2b = FIN(63), *sf2w = FIN(64);
  const float *slnb = FIN(65), *slng = FIN(66);
  #undef FIN

  float* ws = (float*)d_ws;
  size_t off = 0;
  auto alloc = [&](size_t n) { float* p = ws + off; off += (n + 63) & ~(size_t)63; return p; };
  EncBufs eb;
  eb.q = alloc((size_t)BL * DKH);
  eb.k = alloc((size_t)BL * DKH);
  eb.v = alloc((size_t)BL * DKH);
  eb.scores = alloc((size_t)Bn * Hh * Ln * Ln);
  eb.ctx = alloc((size_t)BL * DKH);
  eb.ypre = alloc((size_t)BL * Dm);
  eb.y = alloc((size_t)BL * Dm);
  eb.hffn = alloc((size_t)BL * DKH);
  float* e_out = alloc((size_t)BL * Dm);
  float* s_out = alloc((size_t)BL * Dm);
  float* h1 = alloc((size_t)BL * 128);
  float* h2c = alloc((size_t)BL * 64);
  float* asf = alloc((size_t)BL);
  float* psi = alloc((size_t)BL);
  float* poolw = alloc((size_t)Bn * Ln);
  float* pooled = alloc((size_t)Bn * Dm);
  float* wT = alloc((size_t)128 * DKh * 31);
  float* convfeat = alloc((size_t)BL * 128);
  float* h2 = alloc((size_t)BL * 128);

  hipStream_t st = stream;

  run_encoder(st, enzF, enzD, enzE, enzM, eWqw, eWqb, eWkw, eWkb, eWvw, eWvb, eWow, eWob, ef1w,
              ef1b, ef2w, ef2b, elng, elnb, muDE, sgDE, bDE, muE, sgE, bE, WD1w, WD1b, WD2w, WD2b,
              eb, e_out);
  run_encoder(st, subF, subD, nullptr, subM, sWqw, sWqb, sWkw, sWkb, sWvw, sWvb, sWow, sWob, sf1w,
              sf1b, sf2w, sf2b, slng, slnb, muDS, sgDS, bDS, nullptr, nullptr, nullptr, WD1w, WD1b,
              WD2w, WD2b, eb, s_out);

  gemm_wmma<false, 2><<<ggrid(BL, 128, 1), 128, 0, st>>>(e_out, cls1w, cls1b, nullptr, h1, BL, 128,
                                                         Dm, Dm, 128, 128, 1.f, 1, 0, 0, 0, 0, 0, 0);
  gemm_wmma<false, 2><<<ggrid(BL, 64, 1), 128, 0, st>>>(h1, cls2w, cls2b, nullptr, h2c, BL, 64, 128,
                                                        128, 64, 64, 1.f, 1, 0, 0, 0, 0, 0, 0);
  rowdot_k<0><<<(BL + 255) / 256, 256, 0, st>>>(h2c, cls3w, cls3b, asf, BL, 64);
  act_psi_k<<<(BL + 255) / 256, 256, 0, st>>>(asf, muACT, sgACT, bACT, Wa1w, Wa1b, Wa2w, Wa2b, psi,
                                              BL);
  pool_w_k<<<Bn, 256, 0, st>>>(psi, enzM, poolw);
  pool_e_k<<<(Bn * Dm + 255) / 256, 256, 0, st>>>(poolw, e_out, pooled, Bn * Dm);

  int nwt = 128 * DKh * 31;
  conv_wt_k<<<(nwt + 255) / 256, 256, 0, st>>>(convw, wT, nwt);
  conv1d_k<<<dim3(Ln, Bn), 128, 0, st>>>(s_out, pooled, wT, convb, convfeat);

  gemm_wmma<false, 2><<<ggrid(BL, 128, 1), 128, 0, st>>>(convfeat, fc1w, fc1b, nullptr, h2, BL, 128,
                                                         128, 128, 128, 128, 1.f, 1, 0, 0, 0, 0, 0,
                                                         0);
  rowdot_k<3><<<(BL + 255) / 256, 256, 0, st>>>(h2, fc2w, fc2b, (float*)d_out, BL, 128);
}
